// NERModel_55688545960123
// MI455X (gfx1250) — matrix operand order repaired and checked by reading, not compiled, not run on hardware
//
#include <hip/hip_runtime.h>
#include <hip/hip_bf16.h>
#include <math.h>

// Problem constants (from reference)
#define T_SEQ   2048
#define E_DIM   300
#define E_PAD   320      // pad K to multiple of 32 for WMMA K-loop
#define H_DIM   512
#define H2_DIM  1024     // 2H
#define G_DIM   2048     // 4H (gate dim, PyTorch order i,f,g,o)
#define NTAGS   25
#define NEG_INF -10000.0f

typedef __attribute__((ext_vector_type(16))) __bf16 v16bf;
typedef __attribute__((ext_vector_type(8)))  __bf16 v8bf;
typedef __attribute__((ext_vector_type(8)))  float  v8f;

__device__ __forceinline__ unsigned short f32_to_bf16_bits(float f) {
  unsigned int u = __float_as_uint(f);
  u += 0x7FFFu + ((u >> 16) & 1u);   // round-to-nearest-even
  return (unsigned short)(u >> 16);
}
__device__ __forceinline__ float bf_lo(unsigned int u) { return __uint_as_float(u << 16); }
__device__ __forceinline__ float bf_hi(unsigned int u) { return __uint_as_float(u & 0xFFFF0000u); }

// ---------------------------------------------------------------------------
// 1) Embedding gather -> bf16, padded to E_PAD with zeros.
// ---------------------------------------------------------------------------
__global__ void embed_gather_bf16(const int* __restrict__ sent,
                                  const float* __restrict__ emb,
                                  unsigned short* __restrict__ x_bf) {
  int i = blockIdx.x * blockDim.x + threadIdx.x;
  if (i >= T_SEQ * E_PAD) return;
  int t = i / E_PAD, e = i - t * E_PAD;
  float v = 0.0f;
  if (e < E_DIM) v = emb[(long long)sent[t] * E_DIM + e];
  x_bf[i] = f32_to_bf16_bits(v);
}

// ---------------------------------------------------------------------------
// 2) Generic f32 [rows,k] -> bf16 [rows,kpad] (zero pad).
// ---------------------------------------------------------------------------
__global__ void f32_to_bf16_pad(const float* __restrict__ src,
                                unsigned short* __restrict__ dst,
                                int rows, int k, int kpad) {
  int i = blockIdx.x * blockDim.x + threadIdx.x;
  if (i >= rows * kpad) return;
  int r = i / kpad, c = i - r * kpad;
  float v = (c < k) ? src[(size_t)r * k + c] : 0.0f;
  dst[i] = f32_to_bf16_bits(v);
}

// ---------------------------------------------------------------------------
// 3) WMMA GEMM: C[M,N] = A[M,Kpad] @ W[N,Kpad]^T   (bf16 in, f32 accum/out)
//    One wave per 16x64 output strip: 4 N-tiles share one A fragment
//    (4x A reuse from L2) and produce 4 INDEPENDENT v_wmma chains per
//    K-step (no WMMA->WMMA RAW back-to-back).
//    Fragment layouts used here (contiguous K, same order in A and B):
//      A: lane L holds row M=L%16; K = half*16 + {0..15}
//      B: lane L holds col N=L%16; K = half*16 + {0..15}   (W row-major [N,K]
//         => a B column is a contiguous W row: single 32B load)
//      C: vgpr r holds M = half*8 + r, N = L%16
// ---------------------------------------------------------------------------
__global__ void gemm_bf16_wmma(const unsigned short* __restrict__ A,
                               const unsigned short* __restrict__ W,
                               float* __restrict__ C,
                               int M, int N, int Kpad) {
  const int tile_n0 = blockIdx.x * 4;   // first of 4 N-tiles for this wave
  const int tile_m  = blockIdx.y;
  const int lane = threadIdx.x;
  const int lr   = lane & 15;
  const int half = lane >> 4;

  const __bf16* Arow = (const __bf16*)A + (size_t)(tile_m * 16 + lr) * Kpad;
  const __bf16* Wrow = (const __bf16*)W + (size_t)(tile_n0 * 16 + lr) * Kpad;
  const size_t wstride = (size_t)16 * Kpad;   // next N-tile = +16 W rows

  v8f acc0 = {}, acc1 = {}, acc2 = {}, acc3 = {};
  for (int kb = 0; kb < Kpad; kb += 32) {
    v16bf a = *(const v16bf*)(Arow + kb + half * 16);        // K = kb+half*16+0..15
    const __bf16* wk = Wrow + kb + half * 16;                // K = kb+half*16+0..15
    v16bf b0 = *(const v16bf*)(wk);
    v16bf b1 = *(const v16bf*)(wk + wstride);
    v16bf b2 = *(const v16bf*)(wk + 2 * wstride);
    v16bf b3 = *(const v16bf*)(wk + 3 * wstride);
    acc0 = __builtin_amdgcn_wmma_f32_16x16x32_bf16(false, a, false, b0, (short)0, acc0, false, false);
    acc1 = __builtin_amdgcn_wmma_f32_16x16x32_bf16(false, a, false, b1, (short)0, acc1, false, false);
    acc2 = __builtin_amdgcn_wmma_f32_16x16x32_bf16(false, a, false, b2, (short)0, acc2, false, false);
    acc3 = __builtin_amdgcn_wmma_f32_16x16x32_bf16(false, a, false, b3, (short)0, acc3, false, false);
  }

  float* Cout = C + (size_t)(tile_m * 16 + half * 8) * N + tile_n0 * 16 + lr;
#pragma unroll
  for (int r = 0; r < 8; ++r) {
    Cout[(size_t)r * N]      = acc0[r];
    Cout[(size_t)r * N + 16] = acc1[r];
    Cout[(size_t)r * N + 32] = acc2[r];
    Cout[(size_t)r * N + 48] = acc3[r];
  }
}

// ---------------------------------------------------------------------------
// 4) Persistent BiLSTM recurrence: grid.x = 2 (dir 0 = fwd, 1 = bwd) so the
//    two independent serial chains of a layer run concurrently.
//    1024 threads; each thread owns gate rows {tid, tid+1024} of W_hh.
//    W_hh is pre-converted to bf16 (halves the per-step L2 re-stream from
//    4 MB to 2 MB per direction); dot products in f32. h/c/gates in LDS.
// ---------------------------------------------------------------------------
__global__ void __launch_bounds__(1024)
bilstm_recurrence(const float* __restrict__ pre_f, const float* __restrict__ pre_b,
                  const unsigned short* __restrict__ whhbf_f,
                  const unsigned short* __restrict__ whhbf_b,
                  const float* __restrict__ bih_f, const float* __restrict__ bhh_f,
                  const float* __restrict__ bih_b, const float* __restrict__ bhh_b,
                  float* __restrict__ out) {
  __shared__ float h_sh[H_DIM];
  __shared__ float c_sh[H_DIM];
  __shared__ float gates_sh[G_DIM];

  const int dir = blockIdx.x;
  const int tid = threadIdx.x;
  const float* pre = dir ? pre_b : pre_f;
  const unsigned short* whh = dir ? whhbf_b : whhbf_f;
  const float* bih = dir ? bih_b : bih_f;
  const float* bhh = dir ? bhh_b : bhh_f;
  const float bias0 = bih[tid]        + bhh[tid];
  const float bias1 = bih[tid + 1024] + bhh[tid + 1024];
  const int col_off = dir ? H_DIM : 0;

  if (tid < H_DIM) { h_sh[tid] = 0.0f; c_sh[tid] = 0.0f; }
  __syncthreads();

  const uint4* w0 = (const uint4*)(whh + (size_t)tid * H_DIM);           // 8 bf16/load
  const uint4* w1 = (const uint4*)(whh + (size_t)(tid + 1024) * H_DIM);
  const float4* hv = (const float4*)h_sh;

  for (int it = 0; it < T_SEQ; ++it) {
    const int t = dir ? (T_SEQ - 1 - it) : it;
    const float* prow = pre + (size_t)t * G_DIM;
    float a0 = prow[tid] + bias0;
    float a1 = prow[tid + 1024] + bias1;
#pragma unroll 4
    for (int k = 0; k < H_DIM / 8; ++k) {
      uint4 u0 = w0[k];
      uint4 u1 = w1[k];
      float4 h0 = hv[2 * k];
      float4 h1 = hv[2 * k + 1];
      a0 += bf_lo(u0.x) * h0.x + bf_hi(u0.x) * h0.y +
            bf_lo(u0.y) * h0.z + bf_hi(u0.y) * h0.w +
            bf_lo(u0.z) * h1.x + bf_hi(u0.z) * h1.y +
            bf_lo(u0.w) * h1.z + bf_hi(u0.w) * h1.w;
      a1 += bf_lo(u1.x) * h0.x + bf_hi(u1.x) * h0.y +
            bf_lo(u1.y) * h0.z + bf_hi(u1.y) * h0.w +
            bf_lo(u1.z) * h1.x + bf_hi(u1.z) * h1.y +
            bf_lo(u1.w) * h1.z + bf_hi(u1.w) * h1.w;
    }
    gates_sh[tid]        = a0;
    gates_sh[tid + 1024] = a1;
    __syncthreads();
    if (tid < H_DIM) {
      float ig = 1.0f / (1.0f + __expf(-gates_sh[tid]));
      float fg = 1.0f / (1.0f + __expf(-gates_sh[tid + H_DIM]));
      float gg = tanhf(gates_sh[tid + 2 * H_DIM]);
      float og = 1.0f / (1.0f + __expf(-gates_sh[tid + 3 * H_DIM]));
      float c = fg * c_sh[tid] + ig * gg;
      c_sh[tid] = c;
      float h = og * tanhf(c);
      h_sh[tid] = h;
      out[(size_t)t * H2_DIM + col_off + tid] = h;
    }
    __syncthreads();
  }
}

// ---------------------------------------------------------------------------
// 5) Tag projection: feats[t,tag] = b_tag[tag] + h2[t,:] . w_tag[tag,:]
// ---------------------------------------------------------------------------
__global__ void tag_proj(const float* __restrict__ h2,
                         const float* __restrict__ w_tag,
                         const float* __restrict__ b_tag,
                         float* __restrict__ feats) {
  int i = blockIdx.x * blockDim.x + threadIdx.x;
  if (i >= T_SEQ * NTAGS) return;
  int t = i / NTAGS, tag = i - t * NTAGS;
  const float4* h = (const float4*)(h2 + (size_t)t * H2_DIM);
  const float4* w = (const float4*)(w_tag + (size_t)tag * H2_DIM);
  float acc = b_tag[tag];
  for (int k = 0; k < H2_DIM / 4; ++k) {
    float4 a = h[k], b = w[k];
    acc += a.x * b.x + a.y * b.y + a.z * b.z + a.w * b.w;
  }
  feats[i] = acc;
}

// ---------------------------------------------------------------------------
// 6) Viterbi: single wave, 25 tags. fv0 = [0, -1e4, ...]. Tie-break = first
//    max (matches jnp.argmax). Backtrack writes tag ids as floats to d_out.
// ---------------------------------------------------------------------------
__global__ void viterbi_kernel(const float* __restrict__ feats,
                               const float* __restrict__ trans,
                               int* __restrict__ bp,
                               float* __restrict__ out_path) {
  __shared__ float fv[NTAGS];
  __shared__ float fv_new[NTAGS];
  __shared__ float tr[NTAGS * NTAGS];
  int tid = threadIdx.x;
  for (int i = tid; i < NTAGS * NTAGS; i += blockDim.x) tr[i] = trans[i];
  if (tid < NTAGS) fv[tid] = (tid == 0) ? 0.0f : NEG_INF;
  __syncthreads();

  for (int t = 0; t < T_SEQ; ++t) {
    if (tid < NTAGS) {
      float best = -3.4e38f; int bi = 0;
      for (int p = 0; p < NTAGS; ++p) {
        float s = fv[p] + tr[tid * NTAGS + p];  // transitions[next, prev]
        if (s > best) { best = s; bi = p; }
      }
      bp[t * NTAGS + tid] = bi;
      fv_new[tid] = best + feats[t * NTAGS + tid];
    }
    __syncthreads();
    if (tid < NTAGS) fv[tid] = fv_new[tid];
    __syncthreads();
  }

  if (tid == 0) {
    int best = 0; float bv = fv[0];
    for (int p = 1; p < NTAGS; ++p) if (fv[p] > bv) { bv = fv[p]; best = p; }
    int tag = best;
    for (int t = T_SEQ - 1; t >= 0; --t) {
      out_path[t] = (float)tag;
      tag = bp[t * NTAGS + tag];
    }
  }
}

// ---------------------------------------------------------------------------
// Launch
// ---------------------------------------------------------------------------
extern "C" void kernel_launch(void* const* d_in, const int* in_sizes, int n_in,
                              void* d_out, int out_size, void* d_ws, size_t ws_size,
                              hipStream_t stream) {
  (void)in_sizes; (void)n_in; (void)out_size; (void)ws_size;

  const int*   sent       = (const int*)d_in[0];
  const float* emb        = (const float*)d_in[1];
  const float* w_ih_l0_f  = (const float*)d_in[2];
  const float* w_hh_l0_f  = (const float*)d_in[3];
  const float* b_ih_l0_f  = (const float*)d_in[4];
  const float* b_hh_l0_f  = (const float*)d_in[5];
  const float* w_ih_l0_b  = (const float*)d_in[6];
  const float* w_hh_l0_b  = (const float*)d_in[7];
  const float* b_ih_l0_b  = (const float*)d_in[8];
  const float* b_hh_l0_b  = (const float*)d_in[9];
  const float* w_ih_l1_f  = (const float*)d_in[10];
  const float* w_hh_l1_f  = (const float*)d_in[11];
  const float* b_ih_l1_f  = (const float*)d_in[12];
  const float* b_hh_l1_f  = (const float*)d_in[13];
  const float* w_ih_l1_b  = (const float*)d_in[14];
  const float* w_hh_l1_b  = (const float*)d_in[15];
  const float* b_ih_l1_b  = (const float*)d_in[16];
  const float* b_hh_l1_b  = (const float*)d_in[17];
  const float* w_tag      = (const float*)d_in[18];
  const float* b_tag      = (const float*)d_in[19];
  const float* trans      = (const float*)d_in[20];

  // Workspace carve-up (all offsets 256B-aligned)
  char* ws = (char*)d_ws;
  size_t off = 0;
  auto alloc = [&](size_t bytes) { char* p = ws + off; off += (bytes + 255) & ~(size_t)255; return p; };
  unsigned short* x_bf   = (unsigned short*)alloc((size_t)T_SEQ * E_PAD * 2);   // 1.3 MB
  unsigned short* wA_bf  = (unsigned short*)alloc((size_t)G_DIM * H2_DIM * 2);  // 4 MB (max K)
  unsigned short* wB_bf  = (unsigned short*)alloc((size_t)G_DIM * H2_DIM * 2);  // 4 MB
  unsigned short* whhF_bf = (unsigned short*)alloc((size_t)G_DIM * H_DIM * 2);  // 2 MB
  unsigned short* whhB_bf = (unsigned short*)alloc((size_t)G_DIM * H_DIM * 2);  // 2 MB
  float* pre_f = (float*)alloc((size_t)T_SEQ * G_DIM * 4);                      // 16 MB
  float* pre_b = (float*)alloc((size_t)T_SEQ * G_DIM * 4);                      // 16 MB
  float* h1    = (float*)alloc((size_t)T_SEQ * H2_DIM * 4);                     // 8 MB
  float* h2    = (float*)alloc((size_t)T_SEQ * H2_DIM * 4);                     // 8 MB
  unsigned short* h_bf = (unsigned short*)alloc((size_t)T_SEQ * H2_DIM * 2);    // 4 MB
  float* feats = (float*)alloc((size_t)T_SEQ * NTAGS * 4);
  int*   bp    = (int*)alloc((size_t)T_SEQ * NTAGS * 4);

  const dim3 gemm_grid(G_DIM / 64, T_SEQ / 16);   // 16x64 strip per wave

  // --- Layer 0 ---
  {
    int n = T_SEQ * E_PAD;
    embed_gather_bf16<<<(n + 255) / 256, 256, 0, stream>>>(sent, emb, x_bf);
  }
  {
    int n = G_DIM * E_PAD;
    f32_to_bf16_pad<<<(n + 255) / 256, 256, 0, stream>>>(w_ih_l0_f, wA_bf, G_DIM, E_DIM, E_PAD);
    f32_to_bf16_pad<<<(n + 255) / 256, 256, 0, stream>>>(w_ih_l0_b, wB_bf, G_DIM, E_DIM, E_PAD);
    int nh = G_DIM * H_DIM;
    f32_to_bf16_pad<<<(nh + 255) / 256, 256, 0, stream>>>(w_hh_l0_f, whhF_bf, G_DIM, H_DIM, H_DIM);
    f32_to_bf16_pad<<<(nh + 255) / 256, 256, 0, stream>>>(w_hh_l0_b, whhB_bf, G_DIM, H_DIM, H_DIM);
  }
  gemm_bf16_wmma<<<gemm_grid, 32, 0, stream>>>(x_bf, wA_bf, pre_f, T_SEQ, G_DIM, E_PAD);
  gemm_bf16_wmma<<<gemm_grid, 32, 0, stream>>>(x_bf, wB_bf, pre_b, T_SEQ, G_DIM, E_PAD);
  bilstm_recurrence<<<2, 1024, 0, stream>>>(pre_f, pre_b, whhF_bf, whhB_bf,
                                            b_ih_l0_f, b_hh_l0_f, b_ih_l0_b, b_hh_l0_b, h1);

  // --- Layer 1 ---
  {
    int n = T_SEQ * H2_DIM;
    f32_to_bf16_pad<<<(n + 255) / 256, 256, 0, stream>>>(h1, h_bf, T_SEQ, H2_DIM, H2_DIM);
  }
  {
    int n = G_DIM * H2_DIM;
    f32_to_bf16_pad<<<(n + 255) / 256, 256, 0, stream>>>(w_ih_l1_f, wA_bf, G_DIM, H2_DIM, H2_DIM);
    f32_to_bf16_pad<<<(n + 255) / 256, 256, 0, stream>>>(w_ih_l1_b, wB_bf, G_DIM, H2_DIM, H2_DIM);
    int nh = G_DIM * H_DIM;
    f32_to_bf16_pad<<<(nh + 255) / 256, 256, 0, stream>>>(w_hh_l1_f, whhF_bf, G_DIM, H_DIM, H_DIM);
    f32_to_bf16_pad<<<(nh + 255) / 256, 256, 0, stream>>>(w_hh_l1_b, whhB_bf, G_DIM, H_DIM, H_DIM);
  }
  gemm_bf16_wmma<<<gemm_grid, 32, 0, stream>>>(h_bf, wA_bf, pre_f, T_SEQ, G_DIM, H2_DIM);
  gemm_bf16_wmma<<<gemm_grid, 32, 0, stream>>>(h_bf, wB_bf, pre_b, T_SEQ, G_DIM, H2_DIM);
  bilstm_recurrence<<<2, 1024, 0, stream>>>(pre_f, pre_b, whhF_bf, whhB_bf,
                                            b_ih_l1_f, b_hh_l1_f, b_ih_l1_b, b_hh_l1_b, h2);

  // --- CRF head ---
  {
    int n = T_SEQ * NTAGS;
    tag_proj<<<(n + 255) / 256, 256, 0, stream>>>(h2, w_tag, b_tag, feats);
  }
  viterbi_kernel<<<1, 32, 0, stream>>>(feats, trans, bp, (float*)d_out);
}
